// Voxel_Encoder_13718125543640
// MI455X (gfx1250) — matrix operand order repaired and checked by reading, not compiled, not run on hardware
//
#include <hip/hip_runtime.h>
#include <math.h>

// ---------------------------------------------------------------------------
// Voxel attention encoder for MI455X (gfx1250, wave32, WMMA 16x16x32 f16->f32)
// One workgroup (256 threads = 8 waves) per voxel (B*G = 1024 blocks).
// ---------------------------------------------------------------------------

typedef __attribute__((ext_vector_type(16))) _Float16 v16h;
typedef __attribute__((ext_vector_type(8)))  float    v8f;

union HU { v16h v; _Float16 h[16]; };

// A-operand (16x32 f16, MxK). Row-major A[m][k] in LDS, per-lane K pattern:
// lane = m + 16*half;  VGPR v holds K pair  2v + 8*half + (v>=4 ? 8 : 0)
__device__ __forceinline__ v16h ldA(const _Float16* base, int stride, int m0, int k0) {
  const int lane = threadIdx.x & 31;
  const _Float16* r = base + (m0 + (lane & 15)) * stride + k0 + 8 * (lane >> 4);
  HU x;
#pragma unroll
  for (int v = 0; v < 8; ++v) {
    const int k = 2 * v + (v >= 4 ? 8 : 0);
    x.h[2 * v]     = r[k];
    x.h[2 * v + 1] = r[k + 1];
  }
  return x.v;
}

// B-operand (32x16 f16, KxN) stored TRANSPOSED in LDS as BT[n][k] (row-major).
// lane = n + 16*half; VGPR v holds K pair 2v + 8*half + (v>=4 ? 8 : 0) (matches ldA).
__device__ __forceinline__ v16h ldB(const _Float16* baseT, int stride, int n0, int k0) {
  const int lane = threadIdx.x & 31;
  const _Float16* r = baseT + (n0 + (lane & 15)) * stride + k0 + 8 * (lane >> 4);
  HU x;
#pragma unroll
  for (int v = 0; v < 8; ++v) {
    const int k = 2 * v + (v >= 4 ? 8 : 0);
    x.h[2 * v]     = r[k];
    x.h[2 * v + 1] = r[k + 1];
  }
  return x.v;
}

__device__ __forceinline__ v8f wmma16(v16h a, v16h b, v8f c) {
  return __builtin_amdgcn_wmma_f32_16x16x32_f16(false, a, false, b, (short)0, c,
                                                false, false);
}

#define NPT   32     // points per voxel (K in reference)
#define DIM   128
#define XDIM  384    // DIM * (S+1)
#define LQ    50     // quant table length
#define PCOLS 160    // padded 3*LQ -> multiple of 16

__global__ __launch_bounds__(256) void voxel_encoder_kernel(
    const float* __restrict__ inputs,  const float* __restrict__ coordsg,
    const float* __restrict__ qkv_w,   const float* __restrict__ qkv_b,
    const float* __restrict__ table_x, const float* __restrict__ table_y,
    const float* __restrict__ table_z, const float* __restrict__ trans_w,
    const float* __restrict__ trans_b, const float* __restrict__ ln_g,
    const float* __restrict__ ln_b,    const int* __restrict__ groups,
    float* __restrict__ d_out, int Btot)
{
  // ---- LDS (~223 KB total; one WG per WGP) --------------------------------
  __shared__ _Float16 sh_wbuf[XDIM * DIM];   // 96KB: qkv_w^T [384][128] or trans_w^T [128][384]
  __shared__ _Float16 sh_tbl[PCOLS * DIM];   // 40KB: table^T [160][128] (rows 150..159 = 0)
  __shared__ _Float16 sh_x[NPT * XDIM];      // 24KB: concat [feats | f1 | f2], row stride 384
  __shared__ _Float16 sh_q[NPT * DIM];       //  8KB
  __shared__ _Float16 sh_k[NPT * DIM];       //  8KB (A for P-GEMM, B^T for QK^T)
  __shared__ _Float16 sh_vT[DIM * NPT];      //  8KB: v^T [dim][point]
  __shared__ float    sh_p[NPT * PCOLS];     // 20KB: P scores; reused as h [32][128] at end
  __shared__ float    sh_sc[NPT * NPT];      //  4KB
  __shared__ _Float16 sh_at[NPT * NPT];      //  2KB
  __shared__ unsigned sh_idx[NPT * NPT];     //  4KB packed (ix | iy<<8 | iz<<16)
  __shared__ float    sh_c[NPT * 3];
  __shared__ float    sh_qb[XDIM];
  __shared__ float    sh_tb[DIM], sh_lg[DIM], sh_lb[DIM], sh_pool[DIM];

  const int tid  = threadIdx.x;
  const int wid  = tid >> 5;
  const int lane = tid & 31;
  const int ln16 = lane & 15;
  const int hf   = lane >> 4;

  const int bg = blockIdx.x;
  const int b  = bg >> 9;        // G = 512
  const int g  = bg & 511;
  const int* grp = groups + (size_t)bg * NPT;

  // ---- gather features (float4) / coords / small params ------------------
  for (int e = tid; e < NPT * DIM / 4; e += 256) {
    const int p = e >> 5, d4 = (e & 31) * 4;
    const float4 f4 =
        *(const float4*)(inputs + ((size_t)b * 16384 + grp[p]) * DIM + d4);
    _Float16* dst = sh_x + p * XDIM + d4;
    dst[0] = (_Float16)f4.x; dst[1] = (_Float16)f4.y;
    dst[2] = (_Float16)f4.z; dst[3] = (_Float16)f4.w;
  }
  if (tid < NPT * 3) {
    const int p = tid / 3, a = tid % 3;
    sh_c[tid] = coordsg[((size_t)b * 16384 + grp[p]) * 3 + a];
  }
  if (tid < DIM) { sh_tb[tid] = trans_b[tid]; sh_lg[tid] = ln_g[tid]; sh_lb[tid] = ln_b[tid]; }
  __syncthreads();

  // ---- pairwise quantized distance indices (same for both stages) --------
  for (int e = tid; e < NPT * NPT; e += 256) {
    const int i = e >> 5, j = e & 31;
    unsigned pk = 0;
#pragma unroll
    for (int a = 0; a < 3; ++a) {
      const float dd = sh_c[i * 3 + a] - sh_c[j * 3 + a];
      int q = (int)floorf((dd + 0.25f) / 0.01f);
      q = q < 0 ? 0 : (q > LQ - 1 ? LQ - 1 : q);
      pk |= (unsigned)q << (8 * a);
    }
    sh_idx[e] = pk;
  }

  // ======================= attention stages ================================
  for (int s = 0; s < 2; ++s) {
    // ---- stage weights: qkv_w[s] [128][384] -> wT [384][128] f16 ---------
    const float4* W4 = (const float4*)(qkv_w + (size_t)s * DIM * XDIM);
    for (int e = tid; e < DIM * XDIM / 4; e += 256) {
      const int k = (e * 4) / XDIM, n = (e * 4) % XDIM;
      const float4 w = W4[e];
      sh_wbuf[(n + 0) * DIM + k] = (_Float16)w.x;
      sh_wbuf[(n + 1) * DIM + k] = (_Float16)w.y;
      sh_wbuf[(n + 2) * DIM + k] = (_Float16)w.z;
      sh_wbuf[(n + 3) * DIM + k] = (_Float16)w.w;
    }
    for (int e = tid; e < XDIM; e += 256) sh_qb[e] = qkv_b[s * XDIM + e];
    // ---- tables: tbl^T[l][d] = slab[l][d]; l<50 x, <100 y, <150 z, pad 0 -
    for (int e = tid; e < PCOLS * (DIM / 4); e += 256) {
      const int l = e >> 5, d4 = (e & 31) * 4;
      float4 t;
      if (l < 3 * LQ) {
        const int comp = l / LQ, ll = l % LQ;
        const float* src = (comp == 0 ? table_x : comp == 1 ? table_y : table_z)
                           + ((size_t)(s * 3 + 1) * LQ + ll) * DIM + d4;
        t = *(const float4*)src;
      } else {
        t = make_float4(0.f, 0.f, 0.f, 0.f);
      }
      _Float16* dst = sh_tbl + l * DIM + d4;
      dst[0] = (_Float16)t.x; dst[1] = (_Float16)t.y;
      dst[2] = (_Float16)t.z; dst[3] = (_Float16)t.w;
    }
    __syncthreads();

    // ---- QKV GEMM: [32x128] x [128x384] ----------------------------------
    const _Float16* Af = sh_x + s * DIM;   // stage-s input f (row stride 384)

    // q & k tiles (sections 0,1): 32 tiles; branchless dst pointer select
    for (int t = wid; t < 32; t += 8) {
      const int mt = (t >> 4) & 1, sec = (t >> 3) & 1, nt = t & 7;
      const int nb = sec * 128 + nt * 16;           // column base in [0,256)
      v8f acc = {};
#pragma unroll
      for (int kt = 0; kt < 4; ++kt)
        acc = wmma16(ldA(Af, XDIM, mt * 16, kt * 32),
                     ldB(sh_wbuf, DIM, nb, kt * 32), acc);
      _Float16* dst = sec ? sh_k : sh_q;            // wave-uniform select
      const int lc = nt * 16 + ln16;
      const float bias = sh_qb[nb + ln16];
      const int m0 = mt * 16 + 8 * hf;
#pragma unroll
      for (int r = 0; r < 8; ++r)
        dst[(m0 + r) * DIM + lc] = (_Float16)(acc[r] + bias);
    }
    // v tiles (section 2): 16 tiles; transposed store into vT[dim][point]
    for (int t = wid; t < 16; t += 8) {
      const int mt = t >> 3, nt = t & 7;
      const int nb = 256 + nt * 16;
      v8f acc = {};
#pragma unroll
      for (int kt = 0; kt < 4; ++kt)
        acc = wmma16(ldA(Af, XDIM, mt * 16, kt * 32),
                     ldB(sh_wbuf, DIM, nb, kt * 32), acc);
      const int lc = nt * 16 + ln16;
      const float bias = sh_qb[nb + ln16];
      const int m0 = mt * 16 + 8 * hf;
#pragma unroll
      for (int r = 0; r < 8; ++r)
        sh_vT[lc * NPT + m0 + r] = (_Float16)(acc[r] + bias);
    }
    __syncthreads();

    // ---- P = k . tables^T : [32x128] x [128x160]; 20 tiles ---------------
    for (int t = wid; t < 20; t += 8) {
      const int mt = t & 1, nt = t >> 1;            // nt in 0..9
      v8f acc = {};
#pragma unroll
      for (int kt = 0; kt < 4; ++kt)
        acc = wmma16(ldA(sh_k, DIM, mt * 16, kt * 32),
                     ldB(sh_tbl, DIM, nt * 16, kt * 32), acc);
      const int m0 = mt * 16 + 8 * hf;
      const int cc = nt * 16 + ln16;
#pragma unroll
      for (int r = 0; r < 8; ++r)
        sh_p[(m0 + r) * PCOLS + cc] = acc[r];
    }
    __syncthreads();

    // ---- scores = (q.k^T + bias)/sqrt(d); 4 tiles on waves 0..3 ----------
    if (wid < 4) {
      const int mt = wid >> 1, nt = wid & 1;
      v8f acc = {};
#pragma unroll
      for (int kt = 0; kt < 4; ++kt)
        acc = wmma16(ldA(sh_q, DIM, mt * 16, kt * 32),
                     ldB(sh_k, DIM, nt * 16, kt * 32), acc);
      const int j = nt * 16 + ln16;
      const int m0 = mt * 16 + 8 * hf;
#pragma unroll
      for (int r = 0; r < 8; ++r) {
        const int i = m0 + r;
        const unsigned pk = sh_idx[i * NPT + j];
        const float bias = sh_p[i * PCOLS + (pk & 255)]
                         + sh_p[i * PCOLS + LQ + ((pk >> 8) & 255)]
                         + sh_p[i * PCOLS + 2 * LQ + (pk >> 16)];
        sh_sc[i * NPT + j] = (acc[r] + bias) * 0.088388347648318447f; // 1/sqrt(128)
      }
    }
    __syncthreads();

    // ---- softmax rows (32 rows, threads 0..31) ---------------------------
    if (tid < NPT) {
      float mx = -1e30f;
      for (int j = 0; j < NPT; ++j) mx = fmaxf(mx, sh_sc[tid * NPT + j]);
      float sum = 0.f;
      for (int j = 0; j < NPT; ++j) sum += __expf(sh_sc[tid * NPT + j] - mx);
      const float inv = 1.f / sum;
      for (int j = 0; j < NPT; ++j)
        sh_at[tid * NPT + j] = (_Float16)(__expf(sh_sc[tid * NPT + j] - mx) * inv);
    }
    __syncthreads();

    // ---- f = attn @ v : [32x32] x [32x128]; 16 tiles ---------------------
    for (int t = wid; t < 16; t += 8) {
      const int mt = t >> 3, nt = t & 7;
      v8f acc = {};
      acc = wmma16(ldA(sh_at, NPT, mt * 16, 0),
                   ldB(sh_vT, NPT, nt * 16, 0), acc);
      const int m0 = mt * 16 + 8 * hf;
      const int cc = DIM + s * DIM + nt * 16 + ln16;
#pragma unroll
      for (int r = 0; r < 8; ++r)
        sh_x[(m0 + r) * XDIM + cc] = (_Float16)acc[r];
    }
    __syncthreads();
  }

  // ---- trans: h = xcat @ trans_w + b ; trans_w [384][128] -> wT [128][384]
  {
    const float4* W4 = (const float4*)trans_w;
    for (int e = tid; e < XDIM * DIM / 4; e += 256) {
      const int k = (e * 4) / DIM, n = (e * 4) % DIM;
      const float4 w = W4[e];
      sh_wbuf[(n + 0) * XDIM + k] = (_Float16)w.x;
      sh_wbuf[(n + 1) * XDIM + k] = (_Float16)w.y;
      sh_wbuf[(n + 2) * XDIM + k] = (_Float16)w.z;
      sh_wbuf[(n + 3) * XDIM + k] = (_Float16)w.w;
    }
  }
  __syncthreads();

  for (int t = wid; t < 16; t += 8) {
    const int mt = t >> 3, nt = t & 7;
    v8f acc = {};
#pragma unroll
    for (int kt = 0; kt < 12; ++kt)
      acc = wmma16(ldA(sh_x, XDIM, mt * 16, kt * 32),
                   ldB(sh_wbuf, XDIM, nt * 16, kt * 32), acc);
    const int col = nt * 16 + ln16;
    const float bias = sh_tb[col];
    const int m0 = mt * 16 + 8 * hf;
#pragma unroll
    for (int r = 0; r < 8; ++r)
      sh_p[(m0 + r) * DIM + col] = acc[r] + bias;   // reuse sh_p as h[32][128]
  }
  __syncthreads();

  // ---- LayerNorm + ReLU per point row ------------------------------------
  if (tid < NPT) {
    float mu = 0.f;
    for (int d = 0; d < DIM; ++d) mu += sh_p[tid * DIM + d];
    mu *= (1.f / DIM);
    float var = 0.f;
    for (int d = 0; d < DIM; ++d) {
      const float dv = sh_p[tid * DIM + d] - mu;
      var += dv * dv;
    }
    var *= (1.f / DIM);
    const float rstd = rsqrtf(var + 1e-5f);
    for (int d = 0; d < DIM; ++d) {
      const float h = (sh_p[tid * DIM + d] - mu) * rstd * sh_lg[d] + sh_lb[d];
      sh_p[tid * DIM + d] = fmaxf(h, 0.f);
    }
  }
  __syncthreads();

  // ---- max-pool over points + write permuted grid output -----------------
  if (tid < DIM) {
    float mx = sh_p[tid];
    for (int i = 1; i < NPT; ++i) mx = fmaxf(mx, sh_p[i * DIM + tid]);
    sh_pool[tid] = mx;
    // g = m*64 + n*8 + t  ->  out index og = t*64 + n*8 + m
    const int m = g >> 6, nn = (g >> 3) & 7, tt = g & 7;
    const int og = tt * 64 + nn * 8 + m;
    d_out[((size_t)b * 512 + og) * DIM + tid] = mx;
  }
  __syncthreads();

  // ---- scatter pooled feature back to every point (for_ret) --------------
  float* fr = d_out + (size_t)Btot * 512 * DIM;
  for (int e = tid; e < NPT * DIM; e += 256) {
    const int p = e >> 7, d = e & 127;
    fr[((size_t)b * 16384 + grp[p]) * DIM + d] = sh_pool[d];
  }
}

extern "C" void kernel_launch(void* const* d_in, const int* in_sizes, int n_in,
                              void* d_out, int out_size, void* d_ws, size_t ws_size,
                              hipStream_t stream) {
  (void)n_in; (void)out_size; (void)d_ws; (void)ws_size;
  const float* inputs  = (const float*)d_in[0];
  const float* coords  = (const float*)d_in[1];
  const float* qkv_w   = (const float*)d_in[2];
  const float* qkv_b   = (const float*)d_in[3];
  const float* table_x = (const float*)d_in[4];
  const float* table_y = (const float*)d_in[5];
  const float* table_z = (const float*)d_in[6];
  const float* trans_w = (const float*)d_in[7];
  const float* trans_b = (const float*)d_in[8];
  const float* ln_g    = (const float*)d_in[9];
  const float* ln_b    = (const float*)d_in[10];
  const int*   groups  = (const int*)d_in[11];
  float* out = (float*)d_out;

  const int B = in_sizes[0] / (16384 * 128);
  voxel_encoder_kernel<<<dim3(B * 512), dim3(256), 0, stream>>>(
      inputs, coords, qkv_w, qkv_b, table_x, table_y, table_z,
      trans_w, trans_b, ln_g, ln_b, groups, out, B);
}
